// SemanticSpatialVQ_7335804141733
// MI455X (gfx1250) — matrix operand order repaired and checked by reading, not compiled, not run on hardware
//
#include <hip/hip_runtime.h>
#include <hip/hip_bf16.h>
#include <math.h>

// ---------------- types ----------------
typedef __attribute__((ext_vector_type(16))) __bf16 v16bf;
typedef __attribute__((ext_vector_type(8)))  float  v8f;

#define BN_TOK   36864          // 64*576 tokens
#define DIM      1024
#define NCODES   4096
#define M_BLK    32             // tokens per block
#define PADR     1032           // padded LDS row length in bf16 elems (8-elem pad -> conflict-free)
#define OUT_Q    ((size_t)BN_TOK * DIM)   // 37748736

// LDS budget for the GEMM kernel:
//  Ahi: 32*1032*2 = 66048 B, Alo: 66048 B, scales: 128 B,
//  partial: 1024 B, bestV: 1024 B, bestI: 1024 B  => 135296 B
#define GEMM_SMEM_BYTES 135296

// ---------------- helpers ----------------
__device__ __forceinline__ unsigned short f32_to_bf16_rn(float f) {
  unsigned int u = __float_as_uint(f);
  unsigned int r = u + 0x7FFFu + ((u >> 16) & 1u);   // round-to-nearest-even
  return (unsigned short)(r >> 16);
}
__device__ __forceinline__ float bf16_to_f32(unsigned short h) {
  return __uint_as_float(((unsigned int)h) << 16);
}

union FragU {
  uint4 q[2];
  v16bf v;
};

// A fragment (16x32 bf16, MxK): lane L holds row M=L&15.
// elems 0..15 -> K = (L>>4)*16 + 0..15 (contiguous)
__device__ __forceinline__ v16bf load_a_frag(const unsigned short* p) {
  FragU u;
  u.q[0] = *(const uint4*)(p);
  u.q[1] = *(const uint4*)(p + 8);
  return u.v;
}

// B fragment (32x16 bf16, KxN): lane L holds col N=L&15.
// elems 0..15 -> K = (L>>4)*16 + 0..15 (contiguous)
__device__ __forceinline__ v16bf load_b_frag(const unsigned short* p) {
  FragU u;
  u.q[0] = *(const uint4*)(p);
  u.q[1] = *(const uint4*)(p + 8);
  return u.v;
}

#define WMMA_BF16(A, B, C) \
  __builtin_amdgcn_wmma_f32_16x16x32_bf16(false, (A), false, (B), (short)0, (C), false, false)

// ---------------- kernel 1: codebook normalize + bf16 hi/lo split ----------------
__global__ void __launch_bounds__(256)
vq_codebook_split(const float* __restrict__ codebook,
                  unsigned short* __restrict__ code_hi,
                  unsigned short* __restrict__ code_lo) {
  __shared__ float red[8];
  __shared__ float s_scale;
  const int k = blockIdx.x;
  const int tid = threadIdx.x;

  const float4 v = *((const float4*)(codebook + (size_t)k * DIM) + tid);
  float s = v.x * v.x + v.y * v.y + v.z * v.z + v.w * v.w;
  #pragma unroll
  for (int off = 16; off > 0; off >>= 1) s += __shfl_down(s, off);
  if ((tid & 31) == 0) red[tid >> 5] = s;
  __syncthreads();
  if (tid == 0) {
    float t = 0.f;
    #pragma unroll
    for (int i = 0; i < 8; ++i) t += red[i];
    s_scale = 1.0f / fmaxf(sqrtf(t), 1e-12f);
  }
  __syncthreads();

  const float sc = s_scale;
  float f[4] = {v.x * sc, v.y * sc, v.z * sc, v.w * sc};
  unsigned short h[4], l[4];
  #pragma unroll
  for (int j = 0; j < 4; ++j) {
    h[j] = f32_to_bf16_rn(f[j]);
    l[j] = f32_to_bf16_rn(f[j] - bf16_to_f32(h[j]));
  }
  const size_t o = (size_t)k * DIM + (size_t)tid * 4;
  *(uint2*)(code_hi + o) =
      make_uint2((unsigned)h[0] | ((unsigned)h[1] << 16),
                 (unsigned)h[2] | ((unsigned)h[3] << 16));
  *(uint2*)(code_lo + o) =
      make_uint2((unsigned)l[0] | ((unsigned)l[1] << 16),
                 (unsigned)l[2] | ((unsigned)l[3] << 16));
}

// ---------------- kernel 2: zero stats ----------------
__global__ void vq_zero(float* __restrict__ counts, float* __restrict__ sse) {
  const int i = blockIdx.x * blockDim.x + threadIdx.x;
  if (i < NCODES) counts[i] = 0.f;
  if (i == 0) *sse = 0.f;
}

// ---------------- kernel 3: fused normalize + split-bf16 GEMM + argmax ----------------
__global__ void __launch_bounds__(256, 1)
vq_gemm_argmax(const float* __restrict__ inputs,
               const unsigned short* __restrict__ code_hi,
               const unsigned short* __restrict__ code_lo,
               int* __restrict__ indices) {
  extern __shared__ char smem[];
  unsigned short* Ahi   = (unsigned short*)smem;          // 32*PADR bf16
  unsigned short* Alo   = Ahi + M_BLK * PADR;             // 32*PADR bf16
  float* scales         = (float*)(Alo + M_BLK * PADR);   // 32
  float* partial        = scales + 32;                    // 256
  float* bestV          = partial + 256;                  // 8*32
  int*   bestI          = (int*)(bestV + 256);            // 8*32

  const int tid  = threadIdx.x;
  const int row0 = blockIdx.x * M_BLK;

  // ---- phase 1: per-row sum of squares ----
  {
    const int r  = tid >> 3;
    const int c0 = (tid & 7) * 128;
    const float4* src = (const float4*)(inputs + (size_t)(row0 + r) * DIM + c0);
    float s = 0.f;
    #pragma unroll
    for (int i = 0; i < 32; ++i) {
      float4 v = src[i];
      s += v.x * v.x + v.y * v.y + v.z * v.z + v.w * v.w;
    }
    partial[tid] = s;
  }
  __syncthreads();
  if (tid < 32) {
    float s = 0.f;
    #pragma unroll
    for (int i = 0; i < 8; ++i) s += partial[tid * 8 + i];
    scales[tid] = 1.0f / fmaxf(sqrtf(s), 1e-12f);
  }
  __syncthreads();

  // ---- phase 2: normalize + split to bf16 hi/lo into padded LDS tile ----
  {
    const int r  = tid >> 3;
    const int c0 = (tid & 7) * 128;
    const float sc = scales[r];
    const float4* src = (const float4*)(inputs + (size_t)(row0 + r) * DIM + c0);
    unsigned short* dh = Ahi + r * PADR + c0;
    unsigned short* dl = Alo + r * PADR + c0;
    #pragma unroll
    for (int i = 0; i < 32; ++i) {
      float4 v = src[i];
      float f[4] = {v.x * sc, v.y * sc, v.z * sc, v.w * sc};
      unsigned short h[4], l[4];
      #pragma unroll
      for (int j = 0; j < 4; ++j) {
        h[j] = f32_to_bf16_rn(f[j]);
        l[j] = f32_to_bf16_rn(f[j] - bf16_to_f32(h[j]));
      }
      *(uint2*)(dh + i * 4) =
          make_uint2((unsigned)h[0] | ((unsigned)h[1] << 16),
                     (unsigned)h[2] | ((unsigned)h[3] << 16));
      *(uint2*)(dl + i * 4) =
          make_uint2((unsigned)l[0] | ((unsigned)l[1] << 16),
                     (unsigned)l[2] | ((unsigned)l[3] << 16));
    }
  }
  __syncthreads();

  // ---- phase 3: WMMA sweep over all 4096 codes with fused running argmax ----
  const int wave = tid >> 5;
  const int lane = tid & 31;
  const int ln16 = lane & 15;
  const int hi16 = lane >> 4;
  const int kbA  = hi16 * 16;   // A-fragment K base per lane half
  const int kbB  = hi16 * 16;   // B-fragment K base per lane half

  float bv[16];
  int   bi[16];
  #pragma unroll
  for (int i = 0; i < 16; ++i) { bv[i] = -3.4e38f; bi[i] = 0; }

  const unsigned short* a0h = Ahi + ln16 * PADR + kbA;
  const unsigned short* a0l = Alo + ln16 * PADR + kbA;

  for (int step = 0; step < 32; ++step) {
    // Structural guard: forbid hoisting any LDS/global loads of this body out
    // of the step loop (round-2 showed LICM hoisting the step-invariant A-tile
    // fragments, blowing past the VGPR file and spilling them to scratch).
    asm volatile("" ::: "memory");

    const int n0 = step * 128 + wave * 16;
    const unsigned short* bh = code_hi + (size_t)(n0 + ln16) * DIM + kbB;
    const unsigned short* bl = code_lo + (size_t)(n0 + ln16) * DIM + kbB;

    v8f acc0 = {};
    v8f acc1 = {};

    // Pipeline only the long-latency global B loads; A fragments are cheap LDS
    // reads issued on demand (the scheduler interleaves them across the
    // unrolled iterations). The k-loop keeps a clean 32 trip count so the
    // unroller does unroll-by-4, not full unroll.
    v16bf cBh = load_b_frag(bh);
    v16bf cBl = load_b_frag(bl);

    #pragma unroll 4
    for (int kc = 0; kc < 32; ++kc) {
      const int ko = kc * 32;
      // Unconditional next-chunk B load. For kc==31 this overreads <=64 B past
      // the row; that memory is the next ws region (still allocated) and the
      // loaded values are discarded.
      v16bf nBh = load_b_frag(bh + ko + 32);
      v16bf nBl = load_b_frag(bl + ko + 32);

      v16bf A0h = load_a_frag(a0h + ko);
      v16bf A0l = load_a_frag(a0l + ko);
      v16bf A1h = load_a_frag(a0h + 16 * PADR + ko);
      v16bf A1l = load_a_frag(a0l + 16 * PADR + ko);

      acc0 = WMMA_BF16(A0h, cBh, acc0);
      acc1 = WMMA_BF16(A1h, cBh, acc1);
      acc0 = WMMA_BF16(A0h, cBl, acc0);
      acc1 = WMMA_BF16(A1h, cBl, acc1);
      acc0 = WMMA_BF16(A0l, cBh, acc0);
      acc1 = WMMA_BF16(A1l, cBh, acc1);

      cBh = nBh;
      cBl = nBl;
    }

    if (step < 31) {  // prefetch next N-step's codebook rows (L2-resident)
      __builtin_prefetch(bh + (size_t)128 * DIM, 0, 1);
      __builtin_prefetch(bl + (size_t)128 * DIM, 0, 1);
    }

    const int code = n0 + ln16;
    #pragma unroll
    for (int e = 0; e < 8; ++e) {
      const float v0 = acc0[e];
      if (v0 > bv[e]) { bv[e] = v0; bi[e] = code; }
      const float v1 = acc1[e];
      if (v1 > bv[8 + e]) { bv[8 + e] = v1; bi[8 + e] = code; }
    }
  }

  // ---- phase 4: reduce argmax across the 16 lanes sharing a row ----
  #pragma unroll
  for (int m = 0; m < 2; ++m) {
    #pragma unroll
    for (int e = 0; e < 8; ++e) {
      float v = bv[m * 8 + e];
      int   i = bi[m * 8 + e];
      #pragma unroll
      for (int off = 1; off < 16; off <<= 1) {
        const float ov = __shfl_xor(v, off);
        const int   oi = __shfl_xor(i, off);
        if (ov > v || (ov == v && oi < i)) { v = ov; i = oi; }
      }
      if (ln16 == 0) {
        const int rl = m * 16 + hi16 * 8 + e;   // local row 0..31
        bestV[wave * 32 + rl] = v;
        bestI[wave * 32 + rl] = i;
      }
    }
  }
  __syncthreads();

  // ---- phase 5: reduce across the 8 waves' N-slices, emit index ----
  if (tid < 32) {
    float v = bestV[tid];
    int   i = bestI[tid];
    #pragma unroll
    for (int w = 1; w < 8; ++w) {
      const float ov = bestV[w * 32 + tid];
      const int   oi = bestI[w * 32 + tid];
      if (ov > v || (ov == v && oi < i)) { v = ov; i = oi; }
    }
    indices[row0 + tid] = i;
  }
}

// ---------------- kernel 4: gather + output + MSE + histogram ----------------
__global__ void __launch_bounds__(256)
vq_gather(const float* __restrict__ inputs,
          const float* __restrict__ codebook,
          const int* __restrict__ indices,
          float* __restrict__ out,
          float* __restrict__ counts,
          float* __restrict__ sse) {
  __shared__ float red[8];
  const int t   = blockIdx.x;
  const int tid = threadIdx.x;
  const int idx = indices[t];

  const float4 qv = *((const float4*)(codebook + (size_t)idx * DIM) + tid);
  const float4 xv = *((const float4*)(inputs + (size_t)t * DIM) + tid);
  *((float4*)(out + (size_t)t * DIM) + tid) = qv;  // quantized_st == quantized

  const float dx = qv.x - xv.x, dy = qv.y - xv.y, dz = qv.z - xv.z, dw = qv.w - xv.w;
  float s = dx * dx + dy * dy + dz * dz + dw * dw;
  #pragma unroll
  for (int off = 16; off > 0; off >>= 1) s += __shfl_down(s, off);
  if ((tid & 31) == 0) red[tid >> 5] = s;
  __syncthreads();
  if (tid == 0) {
    float tot = 0.f;
    #pragma unroll
    for (int i = 0; i < 8; ++i) tot += red[i];
    atomicAdd(sse, tot);
    atomicAdd(&counts[idx], 1.0f);
  }
}

// ---------------- kernel 5: losses + perplexity ----------------
__global__ void __launch_bounds__(256)
vq_finalize(const float* __restrict__ counts,
            const float* __restrict__ sse,
            float* __restrict__ out_tail) {
  __shared__ float red[8];
  const int tid = threadIdx.x;
  float s = 0.f;
  for (int i = tid; i < NCODES; i += 256) {
    const float p = counts[i] * (1.0f / (float)BN_TOK);
    s += p * logf(p + 1e-10f);
  }
  #pragma unroll
  for (int off = 16; off > 0; off >>= 1) s += __shfl_down(s, off);
  if ((tid & 31) == 0) red[tid >> 5] = s;
  __syncthreads();
  if (tid == 0) {
    float t = 0.f;
    #pragma unroll
    for (int i = 0; i < 8; ++i) t += red[i];
    // e_latent == q_latent numerically -> vq_loss = 1.25 * MSE
    out_tail[0] = 1.25f * (*sse) * (1.0f / (float)(BN_TOK * (size_t)DIM));
    out_tail[1] = expf(-t);
  }
}

// ---------------- launch ----------------
extern "C" void kernel_launch(void* const* d_in, const int* in_sizes, int n_in,
                              void* d_out, int out_size, void* d_ws, size_t ws_size,
                              hipStream_t stream) {
  const float* inputs   = (const float*)d_in[0];   // [64,576,1024] f32
  const float* codebook = (const float*)d_in[1];   // [4096,1024]   f32
  float* out = (float*)d_out;                      // [36864*1024] quantized + 2 scalars

  // workspace layout (~16.94 MB). NOTE: the GEMM kernel's pipelined B loads
  // may overread <=64 B past code_hi/code_lo ends; the following regions
  // absorb that (values discarded).
  char* ws = (char*)d_ws;
  unsigned short* code_hi = (unsigned short*)ws;                       // 8,388,608 B
  unsigned short* code_lo = code_hi + (size_t)NCODES * DIM;            // 8,388,608 B
  char* p = ws + (size_t)2 * NCODES * DIM * sizeof(unsigned short);
  int*   indices = (int*)p;   p += (size_t)BN_TOK * sizeof(int);       // 147,456 B
  float* counts  = (float*)p; p += (size_t)NCODES * sizeof(float);     // 16,384 B
  float* sse     = (float*)p;                                          // 4 B + pad

  (void)hipFuncSetAttribute((const void*)vq_gemm_argmax,
                            hipFuncAttributeMaxDynamicSharedMemorySize,
                            GEMM_SMEM_BYTES);

  vq_codebook_split<<<NCODES, 256, 0, stream>>>(codebook, code_hi, code_lo);
  vq_zero<<<(NCODES + 256) / 256, 256, 0, stream>>>(counts, sse);
  vq_gemm_argmax<<<BN_TOK / M_BLK, 256, GEMM_SMEM_BYTES, stream>>>(inputs, code_hi, code_lo, indices);
  vq_gather<<<BN_TOK, 256, 0, stream>>>(inputs, codebook, indices, out, counts, sse);
  vq_finalize<<<1, 256, 0, stream>>>(counts, sse, out + OUT_Q);
}
